// EmbedAlign_15685220565642
// MI455X (gfx1250) — matrix operand order repaired and checked by reading, not compiled, not run on hardware
//
#include <hip/hip_runtime.h>
#include <cstdint>
#include <cstddef>

#define DDIM 512
#define LSEQ 1024
#define NNEG 8192
#define G4D  2048   // 4*D
#define BM   128
#define BN   128
#define KT   32

typedef unsigned short u16;
typedef __attribute__((ext_vector_type(16))) __bf16 v16bf;
typedef __attribute__((ext_vector_type(8)))  __bf16 v8bf;
typedef __attribute__((ext_vector_type(8)))  float  v8f;

__device__ __forceinline__ u16 f2bf(float f) {
  unsigned u = __float_as_uint(f);
  u += 0x7FFFu + ((u >> 16) & 1u);           // round-to-nearest-even
  return (u16)(u >> 16);
}
__device__ __forceinline__ float sigmoidf_(float x) { return 1.f / (1.f + __expf(-x)); }

__device__ __forceinline__ float block_reduce_sum(float v) {
  __shared__ float sh[32];
  int lane = threadIdx.x & 31, w = threadIdx.x >> 5;
#pragma unroll
  for (int m = 16; m >= 1; m >>= 1) v += __shfl_xor(v, m, 32);
  if (lane == 0) sh[w] = v;
  __syncthreads();
  int nw = (blockDim.x + 31) >> 5;
  v = (threadIdx.x < (unsigned)nw) ? sh[threadIdx.x] : 0.f;
  if (w == 0) {
#pragma unroll
    for (int m = 16; m >= 1; m >>= 1) v += __shfl_xor(v, m, 32);
  }
  return v;  // valid in thread 0
}

// ---- LDS-relative 32-bit address of a __shared__ pointer ----
__device__ __forceinline__ unsigned lds_addr_of(const u16* p) {
  return (unsigned)(unsigned long long)(__attribute__((address_space(3))) const u16*)p;
}

// ---- CDNA5 async global->LDS DMA (tracked by ASYNCcnt) ----
__device__ __forceinline__ void async_b128(unsigned lds_off, unsigned gbyte_off,
                                           const u16* gbase) {
  asm volatile("global_load_async_to_lds_b128 %0, %1, %2"
               :: "v"(lds_off), "v"(gbyte_off), "s"(gbase)
               : "memory");
}
__device__ __forceinline__ void wait_async0() {
  asm volatile("s_wait_asynccnt 0x0" ::: "memory");
}

// Stage a 128x32 bf16 tile (rows row0..row0+127, cols kb..kb+31 of a row-major
// (rows x ldk) matrix) into LDS. 256 threads x 2 x 16B chunks = 8KB.
__device__ __forceinline__ void issue_tile(const u16* __restrict__ g, int ldk,
                                           int row0, int kb, u16* tile) {
  int tid = threadIdx.x;
#pragma unroll
  for (int i = 0; i < 2; ++i) {
    int c = tid + i * 256;                 // 0..511 chunk id
    int row = c >> 2;                      // 0..127
    int col8 = (c & 3) * 8;                // element offset within tile row
    unsigned goff = (unsigned)(((size_t)(row0 + row) * ldk + kb + col8) * 2);
    async_b128(lds_addr_of(tile + (size_t)c * 8), goff, g);
  }
}

// ---- block-level 128x128 GEMM accumulator, LDS double-buffered ----
// Wave w: rows (w>>1)*32 (2 m-tiles), cols (w&1)*64 (4 n-tiles).
__device__ __forceinline__ void gemm_block_accum(const u16* __restrict__ A,
                                                 const u16* __restrict__ W, int K,
                                                 int mblk, int nblk,
                                                 u16 (*At)[BM * KT], u16 (*Bt)[BN * KT],
                                                 v8f acc[2][4]) {
  int w = threadIdx.x >> 5, lane = threadIdx.x & 31;
  int wm = w >> 1, wn = w & 1;
  int half = lane >> 4, l15 = lane & 15;

  issue_tile(A, K, mblk, 0, At[0]);
  issue_tile(W, K, nblk, 0, Bt[0]);
  wait_async0();
  __syncthreads();

  int buf = 0;
  for (int kb = 0; kb < K; kb += KT, buf ^= 1) {
    if (kb + KT < K) {
      issue_tile(A, K, mblk, kb + KT, At[buf ^ 1]);
      issue_tile(W, K, nblk, kb + KT, Bt[buf ^ 1]);
    }
    const u16* at = At[buf];
    const u16* bt = Bt[buf];
    v16bf areg[2];
#pragma unroll
    for (int mt = 0; mt < 2; ++mt) {
      const u16* ap = at + (size_t)(wm * 32 + mt * 16 + l15) * KT + (half ? 16 : 0);
      v8bf alo = *(const v8bf*)(ap);
      v8bf ahi = *(const v8bf*)(ap + 8);
      areg[mt] = __builtin_shufflevector(alo, ahi, 0,1,2,3,4,5,6,7,8,9,10,11,12,13,14,15);
    }
#pragma unroll
    for (int nt = 0; nt < 4; ++nt) {
      const u16* bp = bt + (size_t)(wn * 64 + nt * 16 + l15) * KT + (half ? 16 : 0);
      v16bf b = *(const v16bf*)bp;
#pragma unroll
      for (int mt = 0; mt < 2; ++mt)
        acc[mt][nt] = __builtin_amdgcn_wmma_f32_16x16x32_bf16(
            false, areg[mt], false, b, (short)0, acc[mt][nt], false, false);
    }
    wait_async0();   // next-buffer DMAs complete (overlapped with the WMMAs above)
    __syncthreads(); // all waves' tiles visible; safe to reuse buffers
  }
}

// ---- GEMM with bias + optional relu: C(M,N) = act(A @ W^T + bias) ----
__global__ void __launch_bounds__(256)
k_gemm_bias(const u16* __restrict__ A, const u16* __restrict__ W,
            const float* __restrict__ bias, float* __restrict__ C,
            int M, int N, int K, int relu) {
  __shared__ __align__(128) u16 At[2][BM * KT];
  __shared__ __align__(128) u16 Bt[2][BN * KT];
  int nb = N >> 7;
  int mblk = (blockIdx.x / nb) << 7, nblk = (blockIdx.x % nb) << 7;
  v8f acc[2][4] = {};
  gemm_block_accum(A, W, K, mblk, nblk, At, Bt, acc);
  int w = threadIdx.x >> 5, lane = threadIdx.x & 31;
  int wm = w >> 1, wn = w & 1, half = lane >> 4, l15 = lane & 15;
#pragma unroll
  for (int mt = 0; mt < 2; ++mt)
#pragma unroll
    for (int nt = 0; nt < 4; ++nt) {
      int col = nblk + wn * 64 + nt * 16 + l15;
      float bv = bias[col];
#pragma unroll
      for (int r = 0; r < 8; ++r) {
        int row = mblk + wm * 32 + mt * 16 + r + half * 8;
        float v = acc[mt][nt][r] + bv;
        if (relu) v = fmaxf(v, 0.f);
        C[(size_t)row * N + col] = v;
      }
    }
}

// ---- fused GEMM + exp + row-sum:  rowacc[i] += sum_j exp(z_i . e_j) ----
__global__ void __launch_bounds__(256)
k_expsum(const u16* __restrict__ A, const u16* __restrict__ W,
         float* __restrict__ rowacc, int M, int N, int K) {
  __shared__ __align__(128) u16 At[2][BM * KT];
  __shared__ __align__(128) u16 Bt[2][BN * KT];
  int nb = N >> 7;
  int mblk = (blockIdx.x / nb) << 7, nblk = (blockIdx.x % nb) << 7;
  v8f acc[2][4] = {};
  gemm_block_accum(A, W, K, mblk, nblk, At, Bt, acc);
  int w = threadIdx.x >> 5, lane = threadIdx.x & 31;
  int wm = w >> 1, half = lane >> 4;
#pragma unroll
  for (int mt = 0; mt < 2; ++mt)
#pragma unroll
    for (int r = 0; r < 8; ++r) {
      float s = 0.f;
#pragma unroll
      for (int nt = 0; nt < 4; ++nt) s += __expf(acc[mt][nt][r]);
#pragma unroll
      for (int m = 1; m < 16; m <<= 1) s += __shfl_xor(s, m, 32);  // within 16-lane half
      if ((lane & 15) == 0)
        atomicAdd(&rowacc[mblk + wm * 32 + mt * 16 + r + half * 8], s);
    }
}

// ---- fused alignment: sum += exp(d)/(exp(d)+negf[row]) over tile ----
__global__ void __launch_bounds__(256)
k_align(const u16* __restrict__ A, const u16* __restrict__ W,
        const float* __restrict__ negf, float* __restrict__ sumacc,
        int M, int N, int K) {
  __shared__ __align__(128) u16 At[2][BM * KT];
  __shared__ __align__(128) u16 Bt[2][BN * KT];
  int nb = N >> 7;
  int mblk = (blockIdx.x / nb) << 7, nblk = (blockIdx.x % nb) << 7;
  v8f acc[2][4] = {};
  gemm_block_accum(A, W, K, mblk, nblk, At, Bt, acc);
  int w = threadIdx.x >> 5, lane = threadIdx.x & 31;
  int wm = w >> 1, half = lane >> 4;
  float s = 0.f;
#pragma unroll
  for (int mt = 0; mt < 2; ++mt)
#pragma unroll
    for (int r = 0; r < 8; ++r) {
      float nf = negf[mblk + wm * 32 + mt * 16 + r + half * 8];
#pragma unroll
      for (int nt = 0; nt < 4; ++nt) {
        float e = __expf(acc[mt][nt][r]);
        s += e / (e + nf);
      }
    }
#pragma unroll
  for (int m = 1; m < 32; m <<= 1) s += __shfl_xor(s, m, 32);
  if (lane == 0) atomicAdd(sumacc, s);
}

// ---- sequential BiLSTM recurrence; blockIdx.x = direction ----
__global__ void __launch_bounds__(1024)
k_lstm(const float* __restrict__ preG, const float* __restrict__ Whh,
       float* __restrict__ Hout, int L) {
  int dir = blockIdx.x;
  const float* Wh = Whh + (size_t)dir * G4D * DDIM;
  const float* pg = preG + dir * G4D;
  __shared__ float h[DDIM], c[DDIM], g[G4D];
  int t = threadIdx.x;
  if (t < DDIM) { h[t] = 0.f; c[t] = 0.f; }
  __syncthreads();
  int lane16 = t & 15;
  int grp = t >> 4;  // 64 half-wave groups, each owns rows grp, grp+64, ...
  for (int s = 0; s < L; ++s) {
    int tt = dir ? (L - 1 - s) : s;
    const float* pr = pg + (size_t)tt * (2 * G4D);
    for (int r = grp; r < G4D; r += 64) {
      const float* w = Wh + (size_t)r * DDIM;
      float acc = 0.f;
#pragma unroll 4
      for (int k = lane16; k < DDIM; k += 16) acc += w[k] * h[k];
#pragma unroll
      for (int m = 1; m < 16; m <<= 1) acc += __shfl_xor(acc, m, 32);
      if (lane16 == 0) g[r] = pr[r] + acc;
    }
    __syncthreads();
    if (t < DDIM) {
      float gi = sigmoidf_(g[t]);
      float gf = sigmoidf_(g[DDIM + t]);
      float gg = tanhf(g[2 * DDIM + t]);
      float go = sigmoidf_(g[3 * DDIM + t]);
      float cn = gf * c[t] + gi * gg;
      c[t] = cn;
      float hn = go * tanhf(cn);
      h[t] = hn;
      Hout[(size_t)tt * (2 * DDIM) + dir * DDIM + t] = hn;
    }
    __syncthreads();
  }
}

// ---- elementwise helpers ----
__global__ void k_zero(float* p, int n) {
  int i = blockIdx.x * blockDim.x + threadIdx.x;
  if (i < n) p[i] = 0.f;
}
__global__ void k_cvt(const float* __restrict__ in, u16* __restrict__ out, int n) {
  int i = blockIdx.x * blockDim.x + threadIdx.x;
  if (i < n) out[i] = f2bf(in[i]);
}
__global__ void k_gather_f32(const float* __restrict__ tab, const int* __restrict__ idx,
                             float* __restrict__ out, int n) {
  int e = blockIdx.x * blockDim.x + threadIdx.x;
  if (e < n) { int r = e >> 9, k = e & 511; out[e] = tab[(size_t)idx[r] * DDIM + k]; }
}
__global__ void k_gather_bf16(const float* __restrict__ tab, const int* __restrict__ idx,
                              u16* __restrict__ out, int n) {
  int e = blockIdx.x * blockDim.x + threadIdx.x;
  if (e < n) { int r = e >> 9, k = e & 511; out[e] = f2bf(tab[(size_t)idx[r] * DDIM + k]); }
}
__global__ void k_sumhalves(const float* __restrict__ H, float* __restrict__ h, int n) {
  int e = blockIdx.x * blockDim.x + threadIdx.x;
  if (e < n) {
    int r = e >> 9, k = e & 511;
    h[e] = H[(size_t)r * (2 * DDIM) + k] + H[(size_t)r * (2 * DDIM) + DDIM + k];
  }
}
// z = mu + eps*softplus(spre); accumulate raw KL sum; emit bf16 z
__global__ void k_zkl(const float* __restrict__ mu, const float* __restrict__ spre,
                      const float* __restrict__ eps, float* __restrict__ z,
                      u16* __restrict__ zb, float* __restrict__ klacc, int n) {
  int i = blockIdx.x * blockDim.x + threadIdx.x;
  float kl = 0.f;
  if (i < n) {
    float m = mu[i], sp = spre[i];
    float sg = log1pf(__expf(sp));
    float zz = m + eps[i] * sg;
    z[i] = zz;
    zb[i] = f2bf(zz);
    kl = sg * sg + m * m - 1.f - 2.f * __logf(sg);
  }
  kl = block_reduce_sum(kl);
  if (threadIdx.x == 0) atomicAdd(klacc, kl);
}
__global__ void k_posdot(const float* __restrict__ z, const float* __restrict__ Etab,
                         const int* __restrict__ idx, float* __restrict__ pos) {
  int i = blockIdx.x;
  const float* zr = z + (size_t)i * DDIM;
  const float* er = Etab + (size_t)idx[i] * DDIM;
  float s = 0.f;
  for (int k = threadIdx.x; k < DDIM; k += blockDim.x) s += zr[k] * er[k];
  s = block_reduce_sum(s);
  if (threadIdx.x == 0) pos[i] = __expf(s);
}
__global__ void k_final(const float* __restrict__ pos, const float* __restrict__ neg,
                        const float* __restrict__ acc, float* __restrict__ out) {
  float s = 0.f;
  for (int i = threadIdx.x; i < LSEQ; i += blockDim.x) {
    float p = pos[i];
    s += p / (p + neg[i]);
  }
  s = block_reduce_sum(s);
  if (threadIdx.x == 0) out[0] = -(s + acc[1] * (1.f / (float)LSEQ) - 0.5f * acc[0]);
}

extern "C" void kernel_launch(void* const* d_in, const int* in_sizes, int n_in,
                              void* d_out, int out_size, void* d_ws, size_t ws_size,
                              hipStream_t stream) {
  (void)in_sizes; (void)n_in; (void)out_size; (void)ws_size;
  const int* en     = (const int*)d_in[0];
  const int* fr     = (const int*)d_in[1];
  const int* en_neg = (const int*)d_in[2];
  const int* fr_neg = (const int*)d_in[3];
  const float* emb  = (const float*)d_in[4];
  const float* Wih[3] = {(const float*)d_in[5], (const float*)d_in[8],  (const float*)d_in[11]};
  const float* Whh[3] = {(const float*)d_in[6], (const float*)d_in[9],  (const float*)d_in[12]};
  const float* bl[3]  = {(const float*)d_in[7], (const float*)d_in[10], (const float*)d_in[13]};
  const float* Wmu1 = (const float*)d_in[14]; const float* bmu1 = (const float*)d_in[15];
  const float* Wmu2 = (const float*)d_in[16]; const float* bmu2 = (const float*)d_in[17];
  const float* Ws1  = (const float*)d_in[18]; const float* bs1  = (const float*)d_in[19];
  const float* Ws2  = (const float*)d_in[20]; const float* bs2  = (const float*)d_in[21];
  const float* E_en = (const float*)d_in[22]; const float* E_fr = (const float*)d_in[23];
  const float* eps  = (const float*)d_in[24];
  float* out = (float*)d_out;

  char* base = (char*)d_ws;
  size_t o = 0;
  auto alloc = [&](size_t bytes) -> char* {
    char* p = base + o;
    o = (o + bytes + 255) & ~(size_t)255;
    return p;
  };
  float* x0   = (float*)alloc((size_t)LSEQ * DDIM * 4);
  u16*   Abf  = (u16*)  alloc((size_t)LSEQ * 2 * DDIM * 2);
  u16*   Wbf  = (u16*)  alloc((size_t)2 * G4D * (2 * DDIM) * 2);
  float* preG = (float*)alloc((size_t)LSEQ * 2 * G4D * 4);
  float* Hout = (float*)alloc((size_t)LSEQ * 2 * DDIM * 4);
  float* hbuf = (float*)alloc((size_t)LSEQ * DDIM * 4);
  float* t1   = (float*)alloc((size_t)LSEQ * DDIM * 4);
  float* mu   = (float*)alloc((size_t)LSEQ * DDIM * 4);
  float* spre = (float*)alloc((size_t)LSEQ * DDIM * 4);
  float* zbuf = (float*)alloc((size_t)LSEQ * DDIM * 4);
  u16*   zb   = (u16*)  alloc((size_t)LSEQ * DDIM * 2);
  u16*   hbf  = (u16*)  alloc((size_t)LSEQ * DDIM * 2);
  u16*   t1bf = (u16*)  alloc((size_t)LSEQ * DDIM * 2);
  u16*   W4bf = (u16*)  alloc((size_t)4 * DDIM * DDIM * 2);
  u16*   Ebf  = (u16*)  alloc((size_t)NNEG * DDIM * 2);
  float* accblk = (float*)alloc((size_t)(3 * LSEQ + 16) * 4);
  float* neg_en = accblk;
  float* negf   = accblk + LSEQ;
  float* pos    = accblk + 2 * LSEQ;
  float* acc    = accblk + 3 * LSEQ;  // [0]=raw KL sum, [1]=alignment sum

  auto gblk = [](int n) { return (n + 255) / 256; };
  auto gemm_blocks = [](int M, int N) { return (M >> 7) * (N >> 7); };

  k_zero<<<gblk(3 * LSEQ + 16), 256, 0, stream>>>(accblk, 3 * LSEQ + 16);
  k_gather_f32<<<gblk(LSEQ * DDIM), 256, 0, stream>>>(emb, en, x0, LSEQ * DDIM);
  k_cvt<<<gblk(LSEQ * DDIM), 256, 0, stream>>>(x0, Abf, LSEQ * DDIM);

  const int inK[3] = {DDIM, 2 * DDIM, 2 * DDIM};
  for (int l = 0; l < 3; ++l) {
    int K = inK[l];
    int wn = 2 * G4D * K;
    k_cvt<<<gblk(wn), 256, 0, stream>>>(Wih[l], Wbf, wn);
    int N = 2 * G4D;  // both directions in one GEMM, N = 4096
    k_gemm_bias<<<gemm_blocks(LSEQ, N), 256, 0, stream>>>(Abf, Wbf, bl[l], preG, LSEQ, N, K, 0);
    k_lstm<<<2, 1024, 0, stream>>>(preG, Whh[l], Hout, LSEQ);
    if (l < 2) k_cvt<<<gblk(LSEQ * 2 * DDIM), 256, 0, stream>>>(Hout, Abf, LSEQ * 2 * DDIM);
  }
  k_sumhalves<<<gblk(LSEQ * DDIM), 256, 0, stream>>>(Hout, hbuf, LSEQ * DDIM);
  k_cvt<<<gblk(LSEQ * DDIM), 256, 0, stream>>>(hbuf, hbf, LSEQ * DDIM);

  int wsz = DDIM * DDIM;
  k_cvt<<<gblk(wsz), 256, 0, stream>>>(Wmu1, W4bf + 0 * (size_t)wsz, wsz);
  k_cvt<<<gblk(wsz), 256, 0, stream>>>(Wmu2, W4bf + 1 * (size_t)wsz, wsz);
  k_cvt<<<gblk(wsz), 256, 0, stream>>>(Ws1,  W4bf + 2 * (size_t)wsz, wsz);
  k_cvt<<<gblk(wsz), 256, 0, stream>>>(Ws2,  W4bf + 3 * (size_t)wsz, wsz);

  int gb512 = gemm_blocks(LSEQ, DDIM);
  k_gemm_bias<<<gb512, 256, 0, stream>>>(hbf,  W4bf + 0 * (size_t)wsz, bmu1, t1,  LSEQ, DDIM, DDIM, 1);
  k_cvt<<<gblk(LSEQ * DDIM), 256, 0, stream>>>(t1, t1bf, LSEQ * DDIM);
  k_gemm_bias<<<gb512, 256, 0, stream>>>(t1bf, W4bf + 1 * (size_t)wsz, bmu2, mu,  LSEQ, DDIM, DDIM, 0);
  k_gemm_bias<<<gb512, 256, 0, stream>>>(hbf,  W4bf + 2 * (size_t)wsz, bs1,  t1,  LSEQ, DDIM, DDIM, 1);
  k_cvt<<<gblk(LSEQ * DDIM), 256, 0, stream>>>(t1, t1bf, LSEQ * DDIM);
  k_gemm_bias<<<gb512, 256, 0, stream>>>(t1bf, W4bf + 3 * (size_t)wsz, bs2,  spre, LSEQ, DDIM, DDIM, 0);

  k_zkl<<<gblk(LSEQ * DDIM), 256, 0, stream>>>(mu, spre, eps, zbuf, zb, acc + 0, LSEQ * DDIM);
  k_posdot<<<LSEQ, 256, 0, stream>>>(zbuf, E_en, en, pos);

  k_gather_bf16<<<gblk(NNEG * DDIM), 256, 0, stream>>>(E_en, en_neg, Ebf, NNEG * DDIM);
  k_expsum<<<gemm_blocks(LSEQ, NNEG), 256, 0, stream>>>(zb, Ebf, neg_en, LSEQ, NNEG, DDIM);
  k_gather_bf16<<<gblk(NNEG * DDIM), 256, 0, stream>>>(E_fr, fr_neg, Ebf, NNEG * DDIM);
  k_expsum<<<gemm_blocks(LSEQ, NNEG), 256, 0, stream>>>(zb, Ebf, negf, LSEQ, NNEG, DDIM);

  k_gather_bf16<<<gblk(LSEQ * DDIM), 256, 0, stream>>>(E_fr, fr, Ebf, LSEQ * DDIM);
  k_align<<<gemm_blocks(LSEQ, LSEQ), 256, 0, stream>>>(zb, Ebf, negf, acc + 1, LSEQ, LSEQ, DDIM);

  k_final<<<1, 256, 0, stream>>>(pos, neg_en, acc, out);
}
